// DVGMambaModel_55301998903727
// MI455X (gfx1250) — compile-verified, matrix operand order checked by reading, not run on hardware
//
#include <hip/hip_runtime.h>
#include <hip/hip_bf16.h>
#include <math.h>

// ---------------------------------------------------------------------------
// DVG Mamba forward for MI455X (gfx1250, wave32, WMMA, async-to-LDS).
// - patch-mean commutes with the patch projection: 20-GFLOP patch GEMM ->
//   77MB streaming reduction (HBM floor ~3.3us at 23.3 TB/s) + tiny GEMM.
// - all GEMM operands are bf16 *in memory*: weights converted once per launch
//   (~7us of bandwidth, then L2-resident), activations emitted as bf16 by
//   their producer kernels. The WMMA inner loop is conversion-free:
//   global_load_async_to_lds_b128 (double-buffered, ASYNCcnt) -> ds_load_b32
//   fragment gather -> v_wmma_f32_16x16x32_bf16.
// ---------------------------------------------------------------------------

#define BATCH  2
#define LSEQ   64
#define NTOK   4            // 1 + NACT
#define SEQ    (LSEQ*NTOK)  // 256
#define DMODEL 512
#define DINNER 1024
#define DSTATE 16
#define NACT   3
#define ADIM   4
#define NLAYER 8

typedef __attribute__((ext_vector_type(16))) __bf16 v16bf;
typedef __attribute__((ext_vector_type(8)))  float  v8f;
typedef unsigned short ushort_t;

__device__ __forceinline__ unsigned short f32_to_bf16(float x) {
  unsigned int u = __float_as_uint(x);
  unsigned int r = (u + 0x7FFFu + ((u >> 16) & 1u)) >> 16;  // RNE
  return (unsigned short)r;
}
__device__ __forceinline__ float bf16_to_f32(unsigned short h) {
  return __uint_as_float((unsigned int)h << 16);
}
__device__ __forceinline__ float siluf(float x) { return x / (1.0f + __expf(-x)); }
__device__ __forceinline__ float softplusf(float x) {
  return (x > 20.0f) ? x : log1pf(__expf(x));
}

// ---------------------------------------------------------------------------
// Async stage of one 64x32 bf16 tile (64B/row, row stride 80B: 16B-aligned
// rows for B128, and gather banks 4*(5m mod 16) are conflict-free).
// 64 rows x 4 chunks = 256 lane-tasks -> 2 async instrs per thread per tile.
// Generic LDS pointer low 32 bits == LDS byte offset (ISA 10.2).
// ---------------------------------------------------------------------------
#define LROWB 40  // ushort row stride (80 bytes)
__device__ __forceinline__ void issue_tile_async(const ushort_t* __restrict__ g,
                                                 int ld, int row0, int k0,
                                                 ushort_t* ldsbase, int tid) {
#pragma unroll
  for (int p = 0; p < 2; ++p) {
    int task = p * 128 + tid;          // 0..255
    int r = task >> 2, q = task & 3;   // row, 16B chunk
    unsigned lds_off = (unsigned)(uintptr_t)(&ldsbase[r * LROWB + q * 8]);
    unsigned long long gaddr =
        (unsigned long long)(uintptr_t)(&g[(size_t)(row0 + r) * ld + k0 + q * 8]);
    asm volatile("global_load_async_to_lds_b128 %0, %1, off"
                 :: "v"(lds_off), "v"(gaddr) : "memory");
  }
}

// ---------------------------------------------------------------------------
// WMMA GEMM: C[M,N] = act(A[M,Kd] @ B[N,Kd]^T + bias), A/B bf16, C f32.
// Optional bf16 side-output for cols < nbf (packed, ld = nbf).
// Block = 128 threads (4 wave32s), 64x64 tile; wave w owns rows [16w,16w+16).
// A-fragment per CDNA5 ISA 7.12.2 (16-bit A 16x32): lane<16 -> K{0..7,16..23}
// paired per VGPR; lane>=16 -> +8. B mirrored (N per lane).
// D: VGPR r -> M=r (lanes 0-15) / r+8 (lanes 16-31), N = lane%16.
// ---------------------------------------------------------------------------
__global__ __launch_bounds__(128)
void gemm_bf16_wmma(const ushort_t* __restrict__ A, int lda,
                    const ushort_t* __restrict__ Bw, int ldb,
                    float* __restrict__ C, int ldc, int Kd,
                    const float* __restrict__ bias, int act,
                    ushort_t* __restrict__ Cbf, int nbf)
{
  __shared__ ushort_t ldsbuf[2][2][64 * LROWB];   // 2 x (A,B) x 5KB = 20,480 B
  const int tid  = threadIdx.x;
  const int lane = tid & 31;
  const int wave = tid >> 5;
  const int m0 = blockIdx.y * 64;
  const int n0 = blockIdx.x * 64;

  v8f acc[4];
  acc[0] = v8f{}; acc[1] = v8f{}; acc[2] = v8f{}; acc[3] = v8f{};

  issue_tile_async(A,  lda, m0, 0, &ldsbuf[0][0][0], tid);
  issue_tile_async(Bw, ldb, n0, 0, &ldsbuf[0][1][0], tid);

  int buf = 0;
  for (int k0 = 0; k0 < Kd; k0 += 32) {
    if (k0 + 32 < Kd) {
      // Pipeline next tile pair; current tile's 4 per-wave async ops retire
      // in-order while the next 4 stay in flight.
      issue_tile_async(A,  lda, m0, k0 + 32, &ldsbuf[buf ^ 1][0][0], tid);
      issue_tile_async(Bw, ldb, n0, k0 + 32, &ldsbuf[buf ^ 1][1][0], tid);
      asm volatile("s_wait_asynccnt 0x4" ::: "memory");
    } else {
      asm volatile("s_wait_asynccnt 0x0" ::: "memory");
    }
    __syncthreads();

    const ushort_t* ldsA = &ldsbuf[buf][0][0];
    const ushort_t* ldsB = &ldsbuf[buf][1][0];
    const int khi = (lane & 16) ? 8 : 0;
    const int mn  = lane & 15;

    union { v16bf v; unsigned int u[8]; } af;
    {
      const unsigned int* rowA = (const unsigned int*)&ldsA[(wave * 16 + mn) * LROWB];
#pragma unroll
      for (int v = 0; v < 8; ++v) {
        int kk = ((v < 4) ? (2 * v) : (16 + 2 * (v - 4))) + khi;
        af.u[v] = rowA[kk >> 1];
      }
    }
#pragma unroll
    for (int nt = 0; nt < 4; ++nt) {
      union { v16bf v; unsigned int u[8]; } bfr;
      const unsigned int* rowB = (const unsigned int*)&ldsB[(nt * 16 + mn) * LROWB];
#pragma unroll
      for (int v = 0; v < 8; ++v) {
        int kk = ((v < 4) ? (2 * v) : (16 + 2 * (v - 4))) + khi;
        bfr.u[v] = rowB[kk >> 1];
      }
      acc[nt] = __builtin_amdgcn_wmma_f32_16x16x32_bf16(
          false, af.v, false, bfr.v, (short)0, acc[nt], false, false);
    }
    __syncthreads();   // all waves done reading buf before it is refilled
    buf ^= 1;
  }

  const int rofs = (lane & 16) ? 8 : 0;
#pragma unroll
  for (int nt = 0; nt < 4; ++nt) {
    union { v8f v; float f[8]; } dv; dv.v = acc[nt];
    int col = n0 + nt * 16 + (lane & 15);
    float bv = bias ? bias[col] : 0.0f;
#pragma unroll
    for (int r = 0; r < 8; ++r) {
      int row = m0 + wave * 16 + r + rofs;
      float v = dv.f[r] + bv;
      if (act == 1) v = softplusf(v);
      C[(size_t)row * ldc + col] = v;
      if (Cbf && col < nbf) Cbf[(size_t)row * nbf + col] = f32_to_bf16(v);
    }
  }
}

// --- bulk f32 -> bf16 (weights), 4 elements/thread ------------------------
__global__ __launch_bounds__(256)
void k_cvt_bf16(const float* __restrict__ in, ushort_t* __restrict__ out, int n4) {
  int i = blockIdx.x * 256 + threadIdx.x;
  if (i < n4) {
    float4 f = ((const float4*)in)[i];
    unsigned int lo = (unsigned)f32_to_bf16(f.x) | ((unsigned)f32_to_bf16(f.y) << 16);
    unsigned int hi = (unsigned)f32_to_bf16(f.z) | ((unsigned)f32_to_bf16(f.w) << 16);
    ((uint2*)out)[i] = make_uint2(lo, hi);
  }
}

// --- image patch-mean (bf16 out): f = c*256+py*16+px ----------------------
__global__ __launch_bounds__(256)
void k_img_mean(const float* __restrict__ img, ushort_t* __restrict__ meanp) {
  int bl = blockIdx.x;                          // 0..127
  int f  = blockIdx.y * 256 + threadIdx.x;      // 0..767
  int c = f >> 8, py = (f >> 4) & 15, px = f & 15;
  const float* base = img + ((size_t)bl * 3 + c) * 224 * 224;
  float s = 0.0f;
  for (int Py = 0; Py < 14; ++Py)
    for (int Px = 0; Px < 14; ++Px)
      s += base[(Py * 16 + py) * 224 + (Px * 16 + px)];
  meanp[(size_t)bl * 768 + f] = f32_to_bf16(s * (1.0f / 196.0f));
}

// --- assemble token sequence: [img, state, act0, act1] per timestep -------
__global__ __launch_bounds__(512)
void k_tokens(const float* __restrict__ imgtok,
              const float* __restrict__ states, const float* __restrict__ state_w,
              const float* __restrict__ state_b,
              const float* __restrict__ actions, const float* __restrict__ act_w,
              const float* __restrict__ act_b, float* __restrict__ hidden) {
  int bl = blockIdx.x, ch = threadIdx.x;
  int b = bl >> 6, l = bl & 63;
  float* hrow = hidden + ((size_t)(b * SEQ + l * NTOK)) * DMODEL;
  hrow[ch] = imgtok[(size_t)bl * DMODEL + ch];
  const float* st = states + ((size_t)bl * NACT + 0) * 7;
  float s = state_b[ch];
  for (int j = 0; j < 7; ++j) s += st[j] * state_w[ch * 7 + j];
  hrow[DMODEL + ch] = s;
  for (int t = 0; t < 2; ++t) {
    const float* ac = actions + ((size_t)bl * NACT + t) * ADIM;
    float a = act_b[ch];
    for (int j = 0; j < ADIM; ++j) a += ac[j] * act_w[ch * ADIM + j];
    hrow[(size_t)(2 + t) * DMODEL + ch] = a;
  }
}

// --- residual += hidden; xnorm(bf16) = rms(residual)*w --------------------
__global__ __launch_bounds__(256)
void k_resid_rms(const float* __restrict__ hidden, float* __restrict__ residual,
                 ushort_t* __restrict__ xnorm, const float* __restrict__ w) {
  __shared__ float red[8];
  int row = blockIdx.x, tid = threadIdx.x;
  int lane = tid & 31, wave = tid >> 5;
  size_t base = (size_t)row * DMODEL;
  float v0 = hidden[base + tid]       + residual[base + tid];
  float v1 = hidden[base + 256 + tid] + residual[base + 256 + tid];
  residual[base + tid] = v0;
  residual[base + 256 + tid] = v1;
  float ss = v0 * v0 + v1 * v1;
#pragma unroll
  for (int off = 16; off; off >>= 1) ss += __shfl_xor(ss, off);
  if (lane == 0) red[wave] = ss;
  __syncthreads();
  float tot = 0.0f;
#pragma unroll
  for (int i = 0; i < 8; ++i) tot += red[i];
  float scale = rsqrtf(tot * (1.0f / DMODEL) + 1e-5f);
  xnorm[base + tid]       = f32_to_bf16(v0 * scale * w[tid]);
  xnorm[base + 256 + tid] = f32_to_bf16(v1 * scale * w[256 + tid]);
}

// --- causal depthwise conv (K=4) + SiLU: f32 (for scan) + bf16 (for GEMM) -
__global__ __launch_bounds__(256)
void k_conv_silu(const float* __restrict__ xz, const float* __restrict__ cw,
                 const float* __restrict__ cb, float* __restrict__ xc,
                 ushort_t* __restrict__ xcbf) {
  int row = blockIdx.x;
  int d = blockIdx.y * 256 + threadIdx.x;
  int b = row >> 8, s = row & 255;
  float acc = cb[d];
#pragma unroll
  for (int k = 0; k < 4; ++k) {
    int ss = s + k - 3;
    if (ss >= 0) acc += cw[d * 4 + k] * xz[((size_t)(b * 256 + ss)) * 2048 + d];
  }
  float v = siluf(acc);
  xc[(size_t)row * DINNER + d]   = v;
  xcbf[(size_t)row * DINNER + d] = f32_to_bf16(v);
}

// --- selective scan: one thread per (b,d); bf16 output (feeds out_proj) --
__global__ __launch_bounds__(256)
void k_scan(const float* __restrict__ dtb, const float* __restrict__ xc,
            const float* __restrict__ proj, const float* __restrict__ xz,
            const float* __restrict__ alog, const float* __restrict__ Dp,
            ushort_t* __restrict__ y) {
  int tid = blockIdx.x * 256 + threadIdx.x;   // 0..2047
  int b = tid >> 10, d = tid & 1023;
  float An[DSTATE], h[DSTATE];
  const float* ap = alog + (size_t)d * DSTATE;
#pragma unroll
  for (int n = 0; n < DSTATE; ++n) { An[n] = -__expf(ap[n]); h[n] = 0.0f; }
  float dpv = Dp[d];
  for (int s = 0; s < SEQ; ++s) {
    size_t row = (size_t)b * SEQ + s;
    float dt  = dtb[row * DINNER + d];
    float xcv = xc[row * DINNER + d];
    const float* pr = proj + row * 64;   // [0:32)=dtr, [32:48)=B, [48:64)=C
    float yv = 0.0f;
#pragma unroll
    for (int n = 0; n < DSTATE; ++n) {
      float dA = __expf(dt * An[n]);
      h[n] = dA * h[n] + dt * pr[32 + n] * xcv;
      yv += h[n] * pr[48 + n];
    }
    float z = xz[row * 2048 + DINNER + d];
    y[row * DINNER + d] = f32_to_bf16((yv + dpv * xcv) * siluf(z));
  }
}

// --- MLP gate (bf16 out, feeds fc2) --------------------------------------
__global__ __launch_bounds__(256)
void k_gate(const float* __restrict__ g, ushort_t* __restrict__ out) {
  int row = blockIdx.x;
  int d = blockIdx.y * 256 + threadIdx.x;
  float a = g[(size_t)row * 2048 + d];
  float b = g[(size_t)row * 2048 + DINNER + d];
  out[(size_t)row * DINNER + d] = f32_to_bf16(a * siluf(b));
}

// --- head + per-(b,l) L1: 12 wave32s, one output dot each ----------------
__global__ __launch_bounds__(384)
void k_head(const ushort_t* __restrict__ xnorm, const float* __restrict__ head_w,
            const float* __restrict__ head_b, const float* __restrict__ labels,
            float* __restrict__ out, float* __restrict__ labuf) {
  __shared__ float ssum;
  int bl = blockIdx.x, tid = threadIdx.x;
  int b = bl >> 6, l = bl & 63;
  int wid = tid >> 5, lane = tid & 31;
  if (tid == 0) ssum = 0.0f;
  __syncthreads();
  int t = 1 + (wid >> 2);      // token 1..3
  int a = wid & 3;             // action dim
  const ushort_t* x = xnorm + ((size_t)(b * SEQ + l * NTOK + t)) * DMODEL;
  const float* w = head_w + (size_t)a * DMODEL;
  float s = 0.0f;
  for (int j = lane; j < DMODEL; j += 32) s += bf16_to_f32(x[j]) * w[j];
#pragma unroll
  for (int off = 16; off; off >>= 1) s += __shfl_xor(s, off);
  if (lane == 0) {
    float pred = s + head_b[a];
    int oidx = ((bl)*NACT + (t - 1)) * ADIM + a;
    out[1 + oidx] = pred;
    atomicAdd(&ssum, fabsf(pred - labels[oidx]));
  }
  __syncthreads();
  if (tid == 0) labuf[bl] = ssum * (1.0f / 12.0f);
}

// --- final masked loss ---------------------------------------------------
__global__ __launch_bounds__(128)
void k_loss(const float* __restrict__ labuf, const int* __restrict__ seqlen,
            float* __restrict__ out) {
  __shared__ float bs[2];
  int tid = threadIdx.x;
  if (tid < 2) bs[tid] = 0.0f;
  __syncthreads();
  int b = tid >> 6, l = tid & 63;
  float v = (l < seqlen[b]) ? labuf[tid] : 0.0f;
  atomicAdd(&bs[b], v);
  __syncthreads();
  if (tid == 0) {
    float loss = 0.0f;
    for (int bb = 0; bb < 2; ++bb) {
      int sl = seqlen[bb]; if (sl > 64) sl = 64;
      float cnt = fmaxf((float)sl, 1.0f);
      loss += bs[bb] / cnt;
    }
    out[0] = loss * 0.5f;   // mean over batch, LOSS_COEF = 1
  }
}

__global__ void k_zero(float* p, int n) {
  int i = blockIdx.x * 256 + threadIdx.x;
  if (i < n) p[i] = 0.0f;
}

// ---------------------------------------------------------------------------
extern "C" void kernel_launch(void* const* d_in, const int* in_sizes, int n_in,
                              void* d_out, int out_size, void* d_ws, size_t ws_size,
                              hipStream_t stream) {
  const float* images        = (const float*)d_in[0];
  const int*   seq_length    = (const int*)  d_in[1];
  const float* states        = (const float*)d_in[2];
  const float* actions       = (const float*)d_in[3];
  const float* action_labels = (const float*)d_in[4];
  const float* patch_w       = (const float*)d_in[5];
  const float* patch_b       = (const float*)d_in[6];
  const float* state_w       = (const float*)d_in[7];
  const float* state_b       = (const float*)d_in[8];
  const float* act_w         = (const float*)d_in[9];
  const float* act_b         = (const float*)d_in[10];
  const float* norm1_w       = (const float*)d_in[11];
  const float* in_proj_w     = (const float*)d_in[12];
  const float* conv_w        = (const float*)d_in[13];
  const float* conv_b        = (const float*)d_in[14];
  const float* x_proj_w      = (const float*)d_in[15];
  const float* dt_w          = (const float*)d_in[16];
  const float* dt_b          = (const float*)d_in[17];
  const float* A_log         = (const float*)d_in[18];
  const float* Dp            = (const float*)d_in[19];
  const float* out_proj_w    = (const float*)d_in[20];
  const float* norm2_w       = (const float*)d_in[21];
  const float* fc1_w         = (const float*)d_in[22];
  const float* fc2_w         = (const float*)d_in[23];
  const float* norm_f_w      = (const float*)d_in[24];
  const float* head_w        = (const float*)d_in[25];
  const float* head_b        = (const float*)d_in[26];
  float* out = (float*)d_out;

  // ---- workspace arena (~67 MB): f32 buffers, bf16 activations, bf16 weights
  char* arena = (char*)d_ws;
  auto carve = [&](size_t bytes) {
    char* p = arena;
    arena += (bytes + 255) & ~(size_t)255;
    return p;
  };
  float* hidden   = (float*)carve(262144 * 4);        // 2*256*512
  float* residual = (float*)carve(262144 * 4);
  float* xz       = (float*)carve(1048576 * 4);       // 2*256*2048
  float* xc       = (float*)carve(524288 * 4);        // 2*256*1024
  float* proj     = (float*)carve(32768 * 4);         // 2*256*64
  float* dtb      = (float*)carve(524288 * 4);
  float* imgtok   = (float*)carve(65536 * 4);         // 128*512
  float* labuf    = (float*)carve(128 * 4);
  ushort_t* xnorm_bf = (ushort_t*)carve(262144 * 2);  // 512x512
  ushort_t* xc_bf    = (ushort_t*)carve(524288 * 2);  // 512x1024
  ushort_t* dtr_bf   = (ushort_t*)carve(16384 * 2);   // 512x32
  ushort_t* y_bf     = (ushort_t*)carve(524288 * 2);  // 512x1024
  ushort_t* gate_bf  = (ushort_t*)carve(524288 * 2);  // 512x1024
  ushort_t* meanp_bf = (ushort_t*)carve(98304 * 2);   // 128x768
  ushort_t* patch_wb    = (ushort_t*)carve(393216 * 2);
  ushort_t* in_proj_wb  = (ushort_t*)carve(8388608 * 2);
  ushort_t* x_proj_wb   = (ushort_t*)carve(524288 * 2);
  ushort_t* dt_wb       = (ushort_t*)carve(262144 * 2);
  ushort_t* out_proj_wb = (ushort_t*)carve(4194304 * 2);
  ushort_t* fc1_wb      = (ushort_t*)carve(8388608 * 2);
  ushort_t* fc2_wb      = (ushort_t*)carve(4194304 * 2);

  // ---- one-time weight conversion (L2-resident afterwards) ----
  auto cvt = [&](const float* src, ushort_t* dst, int n) {
    k_cvt_bf16<<<(n / 4 + 255) / 256, 256, 0, stream>>>(src, dst, n / 4);
  };
  cvt(patch_w,    patch_wb,    393216);
  cvt(in_proj_w,  in_proj_wb,  8388608);
  cvt(x_proj_w,   x_proj_wb,   524288);
  cvt(dt_w,       dt_wb,       262144);
  cvt(out_proj_w, out_proj_wb, 4194304);
  cvt(fc1_w,      fc1_wb,      8388608);
  cvt(fc2_w,      fc2_wb,      4194304);

  // ---- tokenize (mean commutes with linear patch projection) ----
  k_img_mean<<<dim3(128, 3), 256, 0, stream>>>(images, meanp_bf);
  gemm_bf16_wmma<<<dim3(DMODEL / 64, 128 / 64), 128, 0, stream>>>(
      meanp_bf, 768, patch_wb, 768, imgtok, DMODEL, 768, patch_b, 0, nullptr, 0);
  k_tokens<<<128, 512, 0, stream>>>(imgtok, states, state_w, state_b,
                                    actions, act_w, act_b, hidden);
  k_zero<<<262144 / 256, 256, 0, stream>>>(residual, 262144);

  const int M = BATCH * SEQ;  // 512 rows
  for (int i = 0; i < NLAYER; ++i) {
    // mamba mixer
    k_resid_rms<<<M, 256, 0, stream>>>(hidden, residual, xnorm_bf, norm1_w + i * DMODEL);
    gemm_bf16_wmma<<<dim3(2048 / 64, M / 64), 128, 0, stream>>>(
        xnorm_bf, DMODEL, in_proj_wb + (size_t)i * 2048 * DMODEL, DMODEL,
        xz, 2048, DMODEL, nullptr, 0, nullptr, 0);
    k_conv_silu<<<dim3(M, 4), 256, 0, stream>>>(
        xz, conv_w + (size_t)i * DINNER * 4, conv_b + (size_t)i * DINNER, xc, xc_bf);
    gemm_bf16_wmma<<<dim3(64 / 64, M / 64), 128, 0, stream>>>(
        xc_bf, DINNER, x_proj_wb + (size_t)i * 64 * DINNER, DINNER,
        proj, 64, DINNER, nullptr, 0, dtr_bf, 32);   // bf16 copy of dtr cols
    gemm_bf16_wmma<<<dim3(DINNER / 64, M / 64), 128, 0, stream>>>(
        dtr_bf, 32, dt_wb + (size_t)i * DINNER * 32, 32,
        dtb, DINNER, 32, dt_b + (size_t)i * DINNER, 1 /*softplus*/, nullptr, 0);
    k_scan<<<(BATCH * DINNER) / 256, 256, 0, stream>>>(
        dtb, xc, proj, xz, A_log + (size_t)i * DINNER * DSTATE,
        Dp + (size_t)i * DINNER, y_bf);
    gemm_bf16_wmma<<<dim3(DMODEL / 64, M / 64), 128, 0, stream>>>(
        y_bf, DINNER, out_proj_wb + (size_t)i * DMODEL * DINNER, DINNER,
        hidden, DMODEL, DINNER, nullptr, 0, nullptr, 0);
    // gated MLP
    k_resid_rms<<<M, 256, 0, stream>>>(hidden, residual, xnorm_bf, norm2_w + i * DMODEL);
    gemm_bf16_wmma<<<dim3(2048 / 64, M / 64), 128, 0, stream>>>(
        xnorm_bf, DMODEL, fc1_wb + (size_t)i * 2048 * DMODEL, DMODEL,
        xz, 2048, DMODEL, nullptr, 0, nullptr, 0);
    k_gate<<<dim3(M, 4), 256, 0, stream>>>(xz, gate_bf);
    gemm_bf16_wmma<<<dim3(DMODEL / 64, M / 64), 128, 0, stream>>>(
        gate_bf, DINNER, fc2_wb + (size_t)i * DMODEL * DINNER, DINNER,
        hidden, DMODEL, DINNER, nullptr, 0, nullptr, 0);
  }

  k_resid_rms<<<M, 256, 0, stream>>>(hidden, residual, xnorm_bf, norm_f_w);
  k_head<<<128, 384, 0, stream>>>(xnorm_bf, head_w, head_b, action_labels, out, labuf);
  k_loss<<<1, 128, 0, stream>>>(labuf, seq_length, out);
}
